// RepirDet_31739808317831
// MI455X (gfx1250) — matrix operand order repaired and checked by reading, not compiled, not run on hardware
//
#include <hip/hip_runtime.h>
#include <cmath>

// ---------------------------------------------------------------------------
// Types / helpers
// ---------------------------------------------------------------------------
typedef __attribute__((ext_vector_type(16))) __bf16 v16bf;
typedef __attribute__((ext_vector_type(8)))  float  v8f;

union V16U { v16bf v; unsigned short u[16]; };

__device__ __forceinline__ unsigned short f2bf(float f) {
    unsigned int u = __float_as_uint(f);
    unsigned int r = u + 0x7FFFu + ((u >> 16) & 1u);   // round-to-nearest-even
    return (unsigned short)(r >> 16);
}
__device__ __forceinline__ float geluf(float x) {
    return 0.5f * x * (1.0f + erff(x * 0.70710678118654752f));
}
__device__ __forceinline__ float sigmf(float x) {
    return 1.0f / (1.0f + expf(-x));
}
// 2x bilinear (half-pixel centers, edge clamp): src coord = y*0.5 - 0.25
__device__ __forceinline__ void up2idx(int y, int H, int& i0, int& i1,
                                       float& w0, float& w1) {
    float s = y * 0.5f - 0.25f;
    int   f = (int)floorf(s);
    w1 = s - (float)f;
    w0 = 1.0f - w1;
    i0 = f < 0 ? 0 : f;
    i1 = (f + 1 > H - 1) ? (H - 1) : (f + 1);
}

// ---------------------------------------------------------------------------
// K_zero: clear GRN sum-of-squares accumulators (512 floats) each call
// ---------------------------------------------------------------------------
__global__ __launch_bounds__(256) void k_zero(float* gxsq) {
    int i = blockIdx.x * 256 + threadIdx.x;
    if (i < 512) gxsq[i] = 0.0f;
}

// ---------------------------------------------------------------------------
// K_thresh: exact 819-th largest logit per batch via 4-pass MSD radix select.
// (sigmoid is monotonic: thresholding logits == thresholding probabilities)
// ---------------------------------------------------------------------------
__global__ __launch_bounds__(256) void k_thresh(const float* __restrict__ logits,
                                                float* __restrict__ thr) {
    __shared__ unsigned int hist[256];
    __shared__ unsigned int sh_prefix;
    __shared__ int          sh_k;
    const int b = blockIdx.x, t = threadIdx.x;
    const float* base = logits + b * 65536;   // (1,256,256), sample [::2,::2]

    unsigned int prefix = 0;
    int k = 819;                              // int(0.05*128*128)
    for (int pass = 0; pass < 4; ++pass) {
        const int shift = 24 - pass * 8;
        const unsigned int pmask =
            (pass == 0) ? 0u : (0xFFFFFFFFu << (shift + 8));
        hist[t] = 0u;
        __syncthreads();
        for (int i = t; i < 16384; i += 256) {
            int h = i >> 7, w = i & 127;
            float f = base[(h * 2) * 256 + (w * 2)];
            unsigned int u = __float_as_uint(f);
            u = (u & 0x80000000u) ? ~u : (u | 0x80000000u); // ascending key
            if ((u & pmask) == prefix)
                atomicAdd(&hist[(u >> shift) & 0xFFu], 1u);
        }
        __syncthreads();
        if (t == 0) {
            int kk = k;
            unsigned int d = 0;
            for (int bin = 255; bin >= 0; --bin) {
                int c = (int)hist[bin];
                if (kk <= c) { d = (unsigned)bin; break; }
                kk -= c;
            }
            sh_prefix = prefix | (d << shift);
            sh_k = kk;
        }
        __syncthreads();
        prefix = sh_prefix;
        k = sh_k;
        __syncthreads();
    }
    if (t == 0) {
        unsigned int u = prefix;
        unsigned int bits = (u & 0x80000000u) ? (u & 0x7FFFFFFFu) : ~u;
        thr[b] = __uint_as_float(bits);
    }
}

// ---------------------------------------------------------------------------
// K_sig: second output = sigmoid(classify_logits), 16*256*256
// ---------------------------------------------------------------------------
__global__ __launch_bounds__(256) void k_sig(const float* __restrict__ logits,
                                             float* __restrict__ out) {
    int i = blockIdx.x * 256 + threadIdx.x;
    out[i] = sigmf(logits[i]);
}

// ---------------------------------------------------------------------------
// K_dwln: depthwise 7x7 conv + mask + channel LayerNorm + mask, fused.
// One thread per pixel; 32 channels held in registers. Output bf16 NHWC.
// ---------------------------------------------------------------------------
__global__ __launch_bounds__(256) void k_dwln(
    const float* __restrict__ feats4x, const float* __restrict__ logits,
    const float* __restrict__ thr, const float* __restrict__ dw_w,
    const float* __restrict__ dw_b, const float* __restrict__ ln_g,
    const float* __restrict__ ln_b, unsigned short* __restrict__ xh) {
    int idx = blockIdx.x * 256 + threadIdx.x;   // 16*128*128
    int w = idx & 127, h = (idx >> 7) & 127, b = idx >> 14;
    float m = (logits[b * 65536 + (h * 2) * 256 + (w * 2)] > thr[b]) ? 1.f : 0.f;

    float vals[32];
    float mu = 0.0f;
#pragma unroll
    for (int c = 0; c < 32; ++c) {
        float acc = dw_b[c];
        const float* ip = feats4x + ((b * 32 + c) << 14);
        for (int ky = 0; ky < 7; ++ky) {
            int iy = h + ky - 3;
            if (iy < 0 || iy > 127) continue;
            for (int kx = 0; kx < 7; ++kx) {
                int ix = w + kx - 3;
                if (ix < 0 || ix > 127) continue;
                acc += ip[(iy << 7) + ix] * dw_w[c * 49 + ky * 7 + kx];
            }
        }
        acc *= m;
        vals[c] = acc;
        mu += acc;
    }
    mu *= (1.0f / 32.0f);
    float var = 0.0f;
#pragma unroll
    for (int c = 0; c < 32; ++c) { float d = vals[c] - mu; var += d * d; }
    var *= (1.0f / 32.0f);
    float rs = rsqrtf(var + 1e-6f);

    unsigned short* op = xh + idx * 32;
#pragma unroll
    for (int c = 0; c < 32; ++c)
        op[c] = f2bf(((vals[c] - mu) * rs * ln_g[c] + ln_b[c]) * m);
}

// ---------------------------------------------------------------------------
// K_pw1: Y1 = gelu(X @ W1^T + b1) with v_wmma_f32_16x16x32_bf16.
// One wave = one 16-pixel x 32-channel tile (two N-halves -> 2 WMMA).
// Also accumulates per-(batch,channel) sum(Y1^2) for GRN.
// ---------------------------------------------------------------------------
__global__ __launch_bounds__(256) void k_pw1(
    const unsigned short* __restrict__ xh, const float* __restrict__ w1,
    const float* __restrict__ b1, unsigned short* __restrict__ y1,
    float* __restrict__ gxsq) {
    __shared__ float acc_sh[32];
    const int lane = threadIdx.x & 31;
    const int wid  = blockIdx.x * 8 + (threadIdx.x >> 5);
    const int pix0 = wid * 16;               // 16 pixels/tile, same batch
    const int bblk = (blockIdx.x * 128) >> 14;

    if (threadIdx.x < 32) acc_sh[threadIdx.x] = 0.0f;
    __syncthreads();

    const int n0 = lane & 15;
    const int hi = lane >> 4;                // lane half
    const int ka = hi * 8;                   // unused
    const int kb = hi * 16;                  // B K-offset

    // A: 16x32 bf16, row m = n0, contiguous K = 16*hi + e
    V16U a;
    const unsigned short* row = xh + (pix0 + n0) * 32;
#pragma unroll
    for (int e = 0; e < 16; ++e)
        a.u[e] = row[e + kb];

    // B: 32x16 bf16, B[k][n] = W1[n][k]; two N-halves
    V16U bv0, bv1;
#pragma unroll
    for (int e = 0; e < 16; ++e) {
        bv0.u[e] = f2bf(w1[n0 * 32 + kb + e]);
        bv1.u[e] = f2bf(w1[(n0 + 16) * 32 + kb + e]);
    }

    v8f c0 = {}, c1 = {};
    c0 = __builtin_amdgcn_wmma_f32_16x16x32_bf16(false, a.v, false, bv0.v,
                                                 (short)0, c0, false, false);
    c1 = __builtin_amdgcn_wmma_f32_16x16x32_bf16(false, a.v, false, bv1.v,
                                                 (short)0, c1, false, false);

    const float bias0 = b1[n0], bias1 = b1[n0 + 16];
    float ss0 = 0.0f, ss1 = 0.0f;
#pragma unroll
    for (int r = 0; r < 8; ++r) {
        int mr = r + hi * 8;
        float ya = geluf(c0[r] + bias0);
        float yb = geluf(c1[r] + bias1);
        ss0 += ya * ya;
        ss1 += yb * yb;
        y1[(pix0 + mr) * 32 + n0]      = f2bf(ya);
        y1[(pix0 + mr) * 32 + n0 + 16] = f2bf(yb);
    }
    atomicAdd(&acc_sh[n0], ss0);
    atomicAdd(&acc_sh[n0 + 16], ss1);
    __syncthreads();
    if (threadIdx.x < 32)
        atomicAdd(&gxsq[bblk * 32 + threadIdx.x], acc_sh[threadIdx.x]);
}

// ---------------------------------------------------------------------------
// K_grn: GRN folded into pw2. scale[b,c] = grn_g[c]*nx[b,c]+1 is folded into
// per-batch bf16 W2eff[n][k] = W2[n][k]*scale[b][k];
// b2eff[b,n] = pw2_b[n] + sum_c grn_b[c]*W2[n][c].
// ---------------------------------------------------------------------------
__global__ void k_grn(const float* __restrict__ gxsq,
                      const float* __restrict__ grn_g,
                      const float* __restrict__ grn_b,
                      const float* __restrict__ pw2_w,
                      const float* __restrict__ pw2_b,
                      unsigned short* __restrict__ w2eff,
                      float* __restrict__ b2eff) {
    __shared__ float sh[32];
    const int b = blockIdx.x, t = threadIdx.x;   // 32 threads
    float g = sqrtf(gxsq[b * 32 + t]);
    sh[t] = g;
    __syncthreads();
    float sum = 0.0f;
    for (int i = 0; i < 32; ++i) sum += sh[i];
    float nx    = g / (sum * (1.0f / 32.0f) + 1e-6f);
    float scale = grn_g[t] * nx + 1.0f;

    float bsum = pw2_b[t];
    for (int n = 0; n < 32; ++n) {
        w2eff[(b * 32 + n) * 32 + t] = f2bf(pw2_w[n * 32 + t] * scale);
        bsum += grn_b[n] * pw2_w[t * 32 + n];
    }
    b2eff[b * 32 + t] = bsum;
}

// ---------------------------------------------------------------------------
// K_pw2: Y3 = Y1 @ W2eff^T + b2eff + shortcut (feats4x). WMMA bf16.
// Writes residual result as bf16 NHWC for the following WMMA conv.
// ---------------------------------------------------------------------------
__global__ __launch_bounds__(256) void k_pw2(
    const unsigned short* __restrict__ y1,
    const unsigned short* __restrict__ w2eff,
    const float* __restrict__ b2eff, const float* __restrict__ feats4x,
    unsigned short* __restrict__ xres) {
    const int lane = threadIdx.x & 31;
    const int wid  = blockIdx.x * 8 + (threadIdx.x >> 5);
    const int pix0 = wid * 16;
    const int b    = pix0 >> 14;

    const int n0 = lane & 15;
    const int hi = lane >> 4;
    const int ka = hi * 8;
    const int kb = hi * 16;

    V16U a;
    const unsigned short* row = y1 + (pix0 + n0) * 32;
#pragma unroll
    for (int e = 0; e < 16; ++e)
        a.u[e] = row[e + kb];

    V16U bv0, bv1;
    const unsigned short* wb = w2eff + b * 1024;
#pragma unroll
    for (int e = 0; e < 16; ++e) {
        bv0.u[e] = wb[n0 * 32 + kb + e];
        bv1.u[e] = wb[(n0 + 16) * 32 + kb + e];
    }

    v8f c0 = {}, c1 = {};
    c0 = __builtin_amdgcn_wmma_f32_16x16x32_bf16(false, a.v, false, bv0.v,
                                                 (short)0, c0, false, false);
    c1 = __builtin_amdgcn_wmma_f32_16x16x32_bf16(false, a.v, false, bv1.v,
                                                 (short)0, c1, false, false);

    const float bias0 = b2eff[b * 32 + n0];
    const float bias1 = b2eff[b * 32 + n0 + 16];
#pragma unroll
    for (int r = 0; r < 8; ++r) {
        int mr  = r + hi * 8;
        int p   = pix0 + mr;
        int loc = p & 16383;
        int h   = loc >> 7, w = loc & 127;
        xres[p * 32 + n0] = f2bf(
            c0[r] + bias0 + feats4x[((b * 32 + n0) << 14) + (h << 7) + w]);
        xres[p * 32 + n0 + 16] = f2bf(
            c1[r] + bias1 + feats4x[((b * 32 + n0 + 16) << 14) + (h << 7) + w]);
    }
}

// ---------------------------------------------------------------------------
// K_tr2: feats2x (16,16,256,256) NCHW f32 -> bf16 NHWC (pixel-major, 16ch)
// ---------------------------------------------------------------------------
__global__ __launch_bounds__(256) void k_tr2(
    const float* __restrict__ feats2x, unsigned short* __restrict__ f2) {
    int idx = blockIdx.x * 256 + threadIdx.x;   // 16*256*256 pixels
    int x = idx & 255, y = (idx >> 8) & 255, b = idx >> 16;
    unsigned short* op = f2 + idx * 16;
#pragma unroll
    for (int c = 0; c < 16; ++c)
        op[c] = f2bf(feats2x[((b * 16 + c) << 16) + (y << 8) + x]);
}

// ---------------------------------------------------------------------------
// K_up_wmma: 3x3 conv (32->8) + BN1 as tap-wise implicit GEMM on WMMA.
// One wave = 16 consecutive pixels of a row; 9 taps, K=32 (ci) each, N=8 of 16.
// ---------------------------------------------------------------------------
__global__ __launch_bounds__(256) void k_up_wmma(
    const unsigned short* __restrict__ xres, const float* __restrict__ up_w,
    const float* __restrict__ up_b, const float* __restrict__ bn_g,
    const float* __restrict__ bn_b, const float* __restrict__ bn_m,
    const float* __restrict__ bn_v, float* __restrict__ up1) {
    const int lane = threadIdx.x & 31;
    const int wid  = blockIdx.x * 8 + (threadIdx.x >> 5);  // 16384 waves
    const int b    = wid >> 10;
    const int rem  = wid & 1023;
    const int y    = rem >> 3;
    const int x0   = (rem & 7) << 4;
    const int n0   = lane & 15, hi = lane >> 4;
    const int ka   = hi * 8, kb = hi * 16;
    const int m    = n0;                        // A row owned by this lane

    v8f acc = {};
#pragma unroll
    for (int t = 0; t < 9; ++t) {
        const int dy = t / 3 - 1, dx = t % 3 - 1;
        const int iy = y + dy;
        if (iy < 0 || iy > 127) continue;       // wave-uniform skip
        int  ix = x0 + m + dx;
        bool vx = (ix >= 0) && (ix <= 127);
        const unsigned short* row =
            xres + (((b << 14) + (iy << 7) + (vx ? ix : 0)) << 5);
        V16U a;
#pragma unroll
        for (int e = 0; e < 16; ++e) {
            int k  = e + kb;
            a.u[e] = vx ? row[k] : (unsigned short)0;
        }
        V16U bv;
#pragma unroll
        for (int e = 0; e < 16; ++e) {
            int k   = kb + e;
            bv.u[e] = (n0 < 8) ? f2bf(up_w[(n0 * 32 + k) * 9 + t])
                               : (unsigned short)0;
        }
        acc = __builtin_amdgcn_wmma_f32_16x16x32_bf16(false, a.v, false, bv.v,
                                                      (short)0, acc, false,
                                                      false);
    }
    const int co = n0 & 7;
    float sc = bn_g[co] * rsqrtf(bn_v[co] + 1e-5f);
    float bs = bn_b[co] + (up_b[co] - bn_m[co]) * sc;
    if (n0 < 8) {
#pragma unroll
        for (int r = 0; r < 8; ++r) {
            int mr = r + hi * 8;
            up1[((b * 8 + n0) << 14) + (y << 7) + (x0 + mr)] = acc[r] * sc + bs;
        }
    }
}

// ---------------------------------------------------------------------------
// K_sh_wmma: 3x3 conv (16->8) + BN2 + ReLU as WMMA implicit GEMM.
// Two taps packed per K=32 (ci=16 each) -> 5 WMMAs. Writes cat channels 8..15.
// ---------------------------------------------------------------------------
__global__ __launch_bounds__(256) void k_sh_wmma(
    const unsigned short* __restrict__ f2, const float* __restrict__ sh_w,
    const float* __restrict__ sh_b, const float* __restrict__ bn_g,
    const float* __restrict__ bn_b, const float* __restrict__ bn_m,
    const float* __restrict__ bn_v, unsigned short* __restrict__ cat) {
    const int lane = threadIdx.x & 31;
    const int wid  = blockIdx.x * 8 + (threadIdx.x >> 5);  // 65536 waves
    const int b    = wid >> 12;
    const int rem  = wid & 4095;
    const int y    = rem >> 4;
    const int x0   = (rem & 15) << 4;
    const int n0   = lane & 15, hi = lane >> 4;
    const int ka   = hi * 8, kb = hi * 16;
    const int m    = n0;

    v8f acc = {};
#pragma unroll
    for (int p = 0; p < 5; ++p) {
        const int t0 = 2 * p, t1 = 2 * p + 1;
        const int dy0 = t0 / 3 - 1, dx0 = t0 % 3 - 1;
        const int dy1 = (t1 < 9) ? (t1 / 3 - 1) : 0;
        const int dx1 = (t1 < 9) ? (t1 % 3 - 1) : 0;
        const int iy0 = y + dy0, iy1 = y + dy1;
        const bool ty0 = (iy0 >= 0) && (iy0 <= 255);
        const bool ty1 = (t1 < 9) && (iy1 >= 0) && (iy1 <= 255);
        if (!ty0 && !ty1) continue;             // wave-uniform skip
        int  ix0 = x0 + m + dx0, ix1 = x0 + m + dx1;
        bool v0  = ty0 && (ix0 >= 0) && (ix0 <= 255);
        bool v1  = ty1 && (ix1 >= 0) && (ix1 <= 255);
        const unsigned short* p0 =
            f2 + (((b << 16) + ((ty0 ? iy0 : 0) << 8) + (v0 ? ix0 : 0)) << 4);
        const unsigned short* p1 =
            f2 + (((b << 16) + ((ty1 ? iy1 : 0) << 8) + (v1 ? ix1 : 0)) << 4);
        V16U a;
#pragma unroll
        for (int e = 0; e < 16; ++e) {
            int k = e + kb;
            a.u[e] = (k < 16) ? (v0 ? p0[k] : (unsigned short)0)
                              : (v1 ? p1[k - 16] : (unsigned short)0);
        }
        const int  tB  = hi ? ((t1 < 9) ? t1 : 0) : t0;
        const bool tBv = hi ? ty1 : ty0;
        V16U bv;
#pragma unroll
        for (int e = 0; e < 16; ++e) {
            bv.u[e] = (n0 < 8 && tBv) ? f2bf(sh_w[(n0 * 16 + e) * 9 + tB])
                                      : (unsigned short)0;
        }
        acc = __builtin_amdgcn_wmma_f32_16x16x32_bf16(false, a.v, false, bv.v,
                                                      (short)0, acc, false,
                                                      false);
    }
    const int co = n0 & 7;
    float sc = bn_g[co] * rsqrtf(bn_v[co] + 1e-5f);
    float bs = bn_b[co] + (sh_b[co] - bn_m[co]) * sc;
    if (n0 < 8) {
#pragma unroll
        for (int r = 0; r < 8; ++r) {
            int   mr = r + hi * 8;
            float v  = acc[r] * sc + bs;
            v = v > 0.0f ? v : 0.0f;
            cat[((b << 16) + (y << 8) + (x0 + mr)) * 16 + 8 + n0] = f2bf(v);
        }
    }
}

// ---------------------------------------------------------------------------
// K_ups: 2x bilinear of up1 (16,8,128,128) -> cat channels 0..7 (bf16 NHWC)
// ---------------------------------------------------------------------------
__global__ __launch_bounds__(256) void k_ups(
    const float* __restrict__ up1, unsigned short* __restrict__ cat) {
    int idx = blockIdx.x * 256 + threadIdx.x;   // 16*256*256 pixels
    int x = idx & 255, y = (idx >> 8) & 255, b = idx >> 16;
    int i0, i1, j0, j1;
    float wy0, wy1, wx0, wx1;
    up2idx(y, 128, i0, i1, wy0, wy1);
    up2idx(x, 128, j0, j1, wx0, wx1);
    unsigned short* op = cat + idx * 16;
#pragma unroll
    for (int c = 0; c < 8; ++c) {
        const float* p = up1 + ((b * 8 + c) << 14);
        float v = wy0 * (wx0 * p[(i0 << 7) + j0] + wx1 * p[(i0 << 7) + j1]) +
                  wy1 * (wx0 * p[(i1 << 7) + j0] + wx1 * p[(i1 << 7) + j1]);
        op[c] = f2bf(v);
    }
}

// ---------------------------------------------------------------------------
// K_last_wmma: 3x3 conv (16->4) over cat (bf16 NHWC), WMMA implicit GEMM,
// two taps per K=32 -> 5 WMMAs. Output lastc f32 NCHW (16,4,256,256).
// ---------------------------------------------------------------------------
__global__ __launch_bounds__(256) void k_last_wmma(
    const unsigned short* __restrict__ cat, const float* __restrict__ last_w,
    const float* __restrict__ last_b, float* __restrict__ lastc) {
    const int lane = threadIdx.x & 31;
    const int wid  = blockIdx.x * 8 + (threadIdx.x >> 5);  // 65536 waves
    const int b    = wid >> 12;
    const int rem  = wid & 4095;
    const int y    = rem >> 4;
    const int x0   = (rem & 15) << 4;
    const int n0   = lane & 15, hi = lane >> 4;
    const int ka   = hi * 8, kb = hi * 16;
    const int m    = n0;
    (void)kb;

    v8f acc = {};
#pragma unroll
    for (int p = 0; p < 5; ++p) {
        const int t0 = 2 * p, t1 = 2 * p + 1;
        const int dy0 = t0 / 3 - 1, dx0 = t0 % 3 - 1;
        const int dy1 = (t1 < 9) ? (t1 / 3 - 1) : 0;
        const int dx1 = (t1 < 9) ? (t1 % 3 - 1) : 0;
        const int iy0 = y + dy0, iy1 = y + dy1;
        const bool ty0 = (iy0 >= 0) && (iy0 <= 255);
        const bool ty1 = (t1 < 9) && (iy1 >= 0) && (iy1 <= 255);
        if (!ty0 && !ty1) continue;
        int  ix0 = x0 + m + dx0, ix1 = x0 + m + dx1;
        bool v0  = ty0 && (ix0 >= 0) && (ix0 <= 255);
        bool v1  = ty1 && (ix1 >= 0) && (ix1 <= 255);
        const unsigned short* p0 =
            cat + (((b << 16) + ((ty0 ? iy0 : 0) << 8) + (v0 ? ix0 : 0)) << 4);
        const unsigned short* p1 =
            cat + (((b << 16) + ((ty1 ? iy1 : 0) << 8) + (v1 ? ix1 : 0)) << 4);
        V16U a;
#pragma unroll
        for (int e = 0; e < 16; ++e) {
            int k = e + kb;
            a.u[e] = (k < 16) ? (v0 ? p0[k] : (unsigned short)0)
                              : (v1 ? p1[k - 16] : (unsigned short)0);
        }
        const int  tB  = hi ? ((t1 < 9) ? t1 : 0) : t0;
        const bool tBv = hi ? ty1 : ty0;
        V16U bv;
#pragma unroll
        for (int e = 0; e < 16; ++e) {
            bv.u[e] = (n0 < 4 && tBv) ? f2bf(last_w[(n0 * 16 + e) * 9 + tB])
                                      : (unsigned short)0;
        }
        acc = __builtin_amdgcn_wmma_f32_16x16x32_bf16(false, a.v, false, bv.v,
                                                      (short)0, acc, false,
                                                      false);
    }
    if (n0 < 4) {
        float bias = last_b[n0 & 3];
#pragma unroll
        for (int r = 0; r < 8; ++r) {
            int mr = r + hi * 8;
            lastc[((b * 4 + n0) << 16) + (y << 8) + (x0 + mr)] = acc[r] + bias;
        }
    }
}

// ---------------------------------------------------------------------------
// K_last2: 5x5 conv (4->1, pad 2) over on-the-fly 2x bilinear of lastc,
// + sigmoid -> first output (16,1,512,512). N=1 -> stays on VALU.
// ---------------------------------------------------------------------------
__global__ __launch_bounds__(256) void k_last2(
    const float* __restrict__ lastc, const float* __restrict__ w5,
    const float* __restrict__ b5, float* __restrict__ out) {
    int idx = blockIdx.x * 256 + threadIdx.x;   // 16*512*512
    int x = idx & 511, y = (idx >> 9) & 511, b = idx >> 18;
    float s = b5[0];
    for (int dy = 0; dy < 5; ++dy) {
        int yy = y + dy - 2;
        if (yy < 0 || yy > 511) continue;
        int i0, i1; float wy0, wy1;
        up2idx(yy, 256, i0, i1, wy0, wy1);
        for (int dx = 0; dx < 5; ++dx) {
            int xx = x + dx - 2;
            if (xx < 0 || xx > 511) continue;
            int j0, j1; float wx0, wx1;
            up2idx(xx, 256, j0, j1, wx0, wx1);
#pragma unroll
            for (int ci = 0; ci < 4; ++ci) {
                const float* p = lastc + ((b * 4 + ci) << 16);
                float v = wy0 * (wx0 * p[(i0 << 8) + j0] +
                                 wx1 * p[(i0 << 8) + j1]) +
                          wy1 * (wx0 * p[(i1 << 8) + j0] +
                                 wx1 * p[(i1 << 8) + j1]);
                s += v * w5[(ci * 5 + dy) * 5 + dx];
            }
        }
    }
    out[idx] = sigmf(s);
}

// ---------------------------------------------------------------------------
// Launch
// ---------------------------------------------------------------------------
extern "C" void kernel_launch(void* const* d_in, const int* in_sizes, int n_in,
                              void* d_out, int out_size, void* d_ws,
                              size_t ws_size, hipStream_t stream) {
    (void)in_sizes; (void)n_in; (void)out_size; (void)ws_size;
    const float* feats4x = (const float*)d_in[0];
    const float* feats2x = (const float*)d_in[1];
    const float* logits  = (const float*)d_in[2];
    const float* dw_w    = (const float*)d_in[3];
    const float* dw_b    = (const float*)d_in[4];
    const float* ln_g    = (const float*)d_in[5];
    const float* ln_b    = (const float*)d_in[6];
    const float* pw1_w   = (const float*)d_in[7];
    const float* pw1_b   = (const float*)d_in[8];
    const float* grn_g   = (const float*)d_in[9];
    const float* grn_b   = (const float*)d_in[10];
    const float* pw2_w   = (const float*)d_in[11];
    const float* pw2_b   = (const float*)d_in[12];
    const float* up_w    = (const float*)d_in[13];
    const float* up_b    = (const float*)d_in[14];
    const float* bn1_g   = (const float*)d_in[15];
    const float* bn1_b   = (const float*)d_in[16];
    const float* bn1_m   = (const float*)d_in[17];
    const float* bn1_v   = (const float*)d_in[18];
    const float* sh_w    = (const float*)d_in[19];
    const float* sh_b    = (const float*)d_in[20];
    const float* bn2_g   = (const float*)d_in[21];
    const float* bn2_b   = (const float*)d_in[22];
    const float* bn2_m   = (const float*)d_in[23];
    const float* bn2_v   = (const float*)d_in[24];
    const float* last_w  = (const float*)d_in[25];
    const float* last_b  = (const float*)d_in[26];
    const float* last2_w = (const float*)d_in[27];
    const float* last2_b = (const float*)d_in[28];

    // Workspace layout with lifetime-based reuse:
    //  [0,       16.78M) xh bf16        -> f2 low half -> lastc f32
    //  [16.78M,  33.55M) y1 bf16        -> f2 high half
    //  [33.55M,  50.33M) xres bf16
    //  [50.33M,  58.72M) up1 f32 (16,8,128,128)
    //  [58.72M,  92.27M) cat bf16 NHWC (16,256,256,16)
    //  [92.27M,  +44KB)  thr / gxsq / w2eff / b2eff
    char* ws = (char*)d_ws;
    unsigned short* xh    = (unsigned short*)(ws);
    unsigned short* y1    = (unsigned short*)(ws + 16777216);
    unsigned short* f2    = (unsigned short*)(ws);           // spans xh+y1
    float*          lastc = (float*)(ws);                    // after f2 dead
    unsigned short* xres  = (unsigned short*)(ws + 33554432);
    float*          up1   = (float*)(ws + 50331648);
    unsigned short* cat   = (unsigned short*)(ws + 58720256);
    float*          thr   = (float*)(ws + 92274688);
    float*          gxsq  = (float*)(ws + 92274688 + 256);
    unsigned short* w2eff = (unsigned short*)(ws + 92274688 + 4096);
    float*          b2eff = (float*)(ws + 92274688 + 40960);

    float* out0 = (float*)d_out;            // sigmoid(out): 16*512*512
    float* out1 = out0 + 16 * 512 * 512;    // sigmoid(classify_logits)

    k_zero     <<<2,     256, 0, stream>>>(gxsq);
    k_thresh   <<<16,    256, 0, stream>>>(logits, thr);
    k_sig      <<<4096,  256, 0, stream>>>(logits, out1);
    k_dwln     <<<1024,  256, 0, stream>>>(feats4x, logits, thr, dw_w, dw_b,
                                           ln_g, ln_b, xh);
    k_pw1      <<<2048,  256, 0, stream>>>(xh, pw1_w, pw1_b, y1, gxsq);
    k_grn      <<<16,    32,  0, stream>>>(gxsq, grn_g, grn_b, pw2_w, pw2_b,
                                           w2eff, b2eff);
    k_pw2      <<<2048,  256, 0, stream>>>(y1, w2eff, b2eff, feats4x, xres);
    k_tr2      <<<4096,  256, 0, stream>>>(feats2x, f2);    // xh,y1 now dead
    k_up_wmma  <<<2048,  256, 0, stream>>>(xres, up_w, up_b, bn1_g, bn1_b,
                                           bn1_m, bn1_v, up1);
    k_ups      <<<4096,  256, 0, stream>>>(up1, cat);       // cat ch 0..7
    k_sh_wmma  <<<8192,  256, 0, stream>>>(f2, sh_w, sh_b, bn2_g, bn2_b,
                                           bn2_m, bn2_v, cat); // cat ch 8..15
    k_last_wmma<<<8192,  256, 0, stream>>>(cat, last_w, last_b, lastc);
    k_last2    <<<16384, 256, 0, stream>>>(lastc, last2_w, last2_b, out0);
}
